// SimplexNeuralODE_58317065945685
// MI455X (gfx1250) — matrix operand order repaired and checked by reading, not compiled, not run on hardware
//
#include <hip/hip_runtime.h>
#include <hip/hip_bf16.h>
#include <math.h>

// ---------------- types ----------------
typedef __attribute__((ext_vector_type(16))) __bf16        v16bf;
typedef __attribute__((ext_vector_type(8)))  float         v8f;
typedef __attribute__((ext_vector_type(8)))  unsigned short ush8;
typedef __attribute__((ext_vector_type(16))) unsigned short ush16;

union FragU {
    ush16 u;
    ush8  h[2];
    v16bf bf;
};

// ---- optional CDNA5 async global->LDS path (guarded; falls back to reg staging) ----
#if defined(__has_builtin)
#if __has_builtin(__builtin_amdgcn_global_load_async_to_lds_b128) && \
    __has_builtin(__builtin_amdgcn_s_wait_asynccnt)
#define HAVE_ASYNC_LDS 1
#endif
#endif

#if defined(HAVE_ASYNC_LDS)
typedef int v4i_vs __attribute__((vector_size(16)));
typedef v4i_vs __attribute__((address_space(1)))* gv4i_p;   // global int4*
typedef v4i_vs __attribute__((address_space(3)))* lv4i_p;   // LDS int4*
__device__ __forceinline__ void async_cp16(const void* g, void* l) {
    __builtin_amdgcn_global_load_async_to_lds_b128(
        (gv4i_p)(unsigned long long)g,
        (lv4i_p)(unsigned int)(unsigned long long)l, 0, 0);
}
#endif

__device__ __forceinline__ unsigned short f2bf(float f) {
    unsigned int u = __float_as_uint(f);
    u = (u + 0x7FFFu + ((u >> 16) & 1u)) >> 16;
    return (unsigned short)u;
}

// branch-free tanh-form GELU (v_exp_f32 + v_rcp_f32, no divergent erf poly)
__device__ __forceinline__ float gelu_f(float x) {
    float u = x * (0.7978845608f + 0.0356774081f * x * x);
    float e = __expf(2.0f * u);
    float t = 1.0f - 2.0f * __builtin_amdgcn_rcpf(e + 1.0f);
    return 0.5f * x * (1.0f + t);
}

// ---------------- conversion kernels ----------------
// activation: (M x K) f32 -> (M x Kp) bf16, zero pad along K
__global__ void cvt_act_pad(const float* __restrict__ src, unsigned short* __restrict__ dst,
                            int M, int K, int Kp) {
    long long idx = (long long)blockIdx.x * blockDim.x + threadIdx.x;
    long long tot = (long long)M * Kp;
    if (idx >= tot) return;
    int r = (int)(idx / Kp), c = (int)(idx % Kp);
    dst[idx] = (c < K) ? f2bf(src[(long long)r * K + c]) : (unsigned short)0;
}

// weight: (K x N) f32 -> TRANSPOSED (N x Kp) bf16, zero pad along K
__global__ void cvt_wT_pad(const float* __restrict__ src, unsigned short* __restrict__ dst,
                           int K, int N, int Kp) {
    long long idx = (long long)blockIdx.x * blockDim.x + threadIdx.x;
    long long tot = (long long)N * Kp;
    if (idx >= tot) return;
    int n = (int)(idx / Kp), k = (int)(idx % Kp);
    dst[idx] = (k < K) ? f2bf(src[(long long)k * N + n]) : (unsigned short)0;
}

// ---------------- main tiled bf16 WMMA GEMM ----------------
// C = act(A(MxK) * Bt(NxK)^T + bias); M%128==0, N%128==0, K%32==0.
// Bt is the weight stored N-major (pre-transposed) so both tiles load contiguously.
// 256 threads = 8 waves in 2(M) x 4(N); each wave computes 64x32 (4x2 tiles of 16x16).
// Double-buffered LDS; async global->LDS copies when available; prefetch of tile k+2.
#define BM 128
#define BN 128
#define BK 32

template<int ACT, bool OUTF, bool OUTB>
__global__ __launch_bounds__(256) void gemm_bf16_wmma(
    const unsigned short* __restrict__ A,
    const unsigned short* __restrict__ Bt,
    const float* __restrict__ bias,
    float* __restrict__ outF,
    unsigned short* __restrict__ outB,
    int M, int N, int K)
{
    __shared__ __align__(16) unsigned short As[2][BM][BK];   // [m][k]
    __shared__ __align__(16) unsigned short Bs[2][BN][BK];   // [n][k]

    const int tid  = threadIdx.x;
    const int wave = tid >> 5;
    const int lane = tid & 31;
    const int wm   = wave >> 2;     // 0..1
    const int wn   = wave & 3;      // 0..3
    const int l15  = lane & 15;
    const int halfsel = lane >> 4;  // 0: lanes 0-15, 1: lanes 16-31
    const int blockM = blockIdx.y * BM;
    const int blockN = blockIdx.x * BN;

    // tile-load assignment: each thread handles one row-half (16 contiguous ushorts)
    const int rL = tid >> 1;            // 0..127
    const int cL = (tid & 1) << 4;      // 0 or 16
    const unsigned short* aBase = A  + (long long)(blockM + rL) * K + cL;
    const unsigned short* bBase = Bt + (long long)(blockN + rL) * K + cL;

    v8f acc[4][2];
#pragma unroll
    for (int i = 0; i < 4; ++i)
#pragma unroll
        for (int j = 0; j < 2; ++j) {
            v8f zv = {0.f,0.f,0.f,0.f,0.f,0.f,0.f,0.f};
            acc[i][j] = zv;
        }

#if defined(HAVE_ASYNC_LDS)
    // ---- async global->LDS double buffering (ASYNCcnt-tracked) ----
    {
        async_cp16(aBase,     &As[0][rL][cL]);
        async_cp16(aBase + 8, &As[0][rL][cL + 8]);
        async_cp16(bBase,     &Bs[0][rL][cL]);
        async_cp16(bBase + 8, &Bs[0][rL][cL + 8]);
    }
    __builtin_amdgcn_s_wait_asynccnt(0);
    __syncthreads();

    int p = 0;
    for (int k0 = 0; k0 < K; k0 += BK, p ^= 1) {
        const bool more = (k0 + BK) < K;
        if (more) {
            async_cp16(aBase + k0 + BK,     &As[p ^ 1][rL][cL]);
            async_cp16(aBase + k0 + BK + 8, &As[p ^ 1][rL][cL + 8]);
            async_cp16(bBase + k0 + BK,     &Bs[p ^ 1][rL][cL]);
            async_cp16(bBase + k0 + BK + 8, &Bs[p ^ 1][rL][cL + 8]);
        }
        if (k0 + 2 * BK < K) {
            __builtin_prefetch(aBase + k0 + 2 * BK, 0, 3);   // global_prefetch_b8
            __builtin_prefetch(bBase + k0 + 2 * BK, 0, 3);
        }

        const int kofA = halfsel << 4;
        const int kofB = halfsel << 4;
        FragU afr[4];
#pragma unroll
        for (int i = 0; i < 4; ++i) {
            int row = wm * 64 + i * 16 + l15;
            afr[i].h[0] = *(const ush8*)&As[p][row][kofA];
            afr[i].h[1] = *(const ush8*)&As[p][row][kofA + 8];
        }
        FragU bfr[2];
#pragma unroll
        for (int j = 0; j < 2; ++j) {
            int n = wn * 32 + j * 16 + l15;
            bfr[j].h[0] = *(const ush8*)&Bs[p][n][kofB];
            bfr[j].h[1] = *(const ush8*)&Bs[p][n][kofB + 8];
        }
#pragma unroll
        for (int i = 0; i < 4; ++i)
#pragma unroll
            for (int j = 0; j < 2; ++j)
                acc[i][j] = __builtin_amdgcn_wmma_f32_16x16x32_bf16(
                    false, afr[i].bf, false, bfr[j].bf,
                    (short)0, acc[i][j], false, false);

        if (more) __builtin_amdgcn_s_wait_asynccnt(0);
        __syncthreads();
    }
#else
    // ---- fallback: register-staged double buffering ----
    ush8 a0 = *(const ush8*)(aBase);
    ush8 a1 = *(const ush8*)(aBase + 8);
    ush8 b0 = *(const ush8*)(bBase);
    ush8 b1 = *(const ush8*)(bBase + 8);
    *(ush8*)&As[0][rL][cL]     = a0;
    *(ush8*)&As[0][rL][cL + 8] = a1;
    *(ush8*)&Bs[0][rL][cL]     = b0;
    *(ush8*)&Bs[0][rL][cL + 8] = b1;
    __syncthreads();

    int p = 0;
    for (int k0 = 0; k0 < K; k0 += BK, p ^= 1) {
        const bool more = (k0 + BK) < K;
        if (more) {
            a0 = *(const ush8*)(aBase + k0 + BK);
            a1 = *(const ush8*)(aBase + k0 + BK + 8);
            b0 = *(const ush8*)(bBase + k0 + BK);
            b1 = *(const ush8*)(bBase + k0 + BK + 8);
        }
        if (k0 + 2 * BK < K) {
            __builtin_prefetch(aBase + k0 + 2 * BK, 0, 3);
            __builtin_prefetch(bBase + k0 + 2 * BK, 0, 3);
        }

        const int kofA = halfsel << 4;
        const int kofB = halfsel << 4;
        FragU afr[4];
#pragma unroll
        for (int i = 0; i < 4; ++i) {
            int row = wm * 64 + i * 16 + l15;
            afr[i].h[0] = *(const ush8*)&As[p][row][kofA];
            afr[i].h[1] = *(const ush8*)&As[p][row][kofA + 8];
        }
        FragU bfr[2];
#pragma unroll
        for (int j = 0; j < 2; ++j) {
            int n = wn * 32 + j * 16 + l15;
            bfr[j].h[0] = *(const ush8*)&Bs[p][n][kofB];
            bfr[j].h[1] = *(const ush8*)&Bs[p][n][kofB + 8];
        }
#pragma unroll
        for (int i = 0; i < 4; ++i)
#pragma unroll
            for (int j = 0; j < 2; ++j)
                acc[i][j] = __builtin_amdgcn_wmma_f32_16x16x32_bf16(
                    false, afr[i].bf, false, bfr[j].bf,
                    (short)0, acc[i][j], false, false);

        if (more) {
            *(ush8*)&As[p ^ 1][rL][cL]     = a0;
            *(ush8*)&As[p ^ 1][rL][cL + 8] = a1;
            *(ush8*)&Bs[p ^ 1][rL][cL]     = b0;
            *(ush8*)&Bs[p ^ 1][rL][cL + 8] = b1;
        }
        __syncthreads();
    }
#endif

    // epilogue: C/D layout: VGPR r -> M = r (lanes 0-15) or r+8 (lanes 16-31)
#pragma unroll
    for (int i = 0; i < 4; ++i)
#pragma unroll
        for (int j = 0; j < 2; ++j) {
            int colBase = blockN + wn * 32 + j * 16 + l15;
            float bv = bias[colBase];
#pragma unroll
            for (int r = 0; r < 8; ++r) {
                int row = blockM + wm * 64 + i * 16 + r + (halfsel << 3);
                float v = acc[i][j][r] + bv;
                if (ACT == 1) v = gelu_f(v);
                long long o = (long long)row * N + colBase;
                if (OUTF) outF[o] = v;
                if (OUTB) outB[o] = f2bf(v);
            }
        }
}

// ---------------- attention (one wave per (b,h), WMMA for both matmuls) ----------------
__global__ __launch_bounds__(32) void attn_kernel(
    const float* __restrict__ qkv,            // (B*T) x 768
    unsigned short* __restrict__ attended)    // (B*T) x 256 bf16
{
    const int bh = blockIdx.x;
    const int b  = bh >> 2;
    const int h  = bh & 3;
    const int lane = threadIdx.x;
    const int l15 = lane & 15;
    const int halfsel = lane >> 4;

    __shared__ __align__(16) unsigned short qs[16][64];
    __shared__ __align__(16) unsigned short ks[16][64];
    __shared__ __align__(16) unsigned short vT[64][32];  // [d][s], s 16..31 zero
    __shared__ float sc[16][16];
    __shared__ __align__(16) unsigned short Ps[16][32];  // probs, K-padded

    for (int idx = lane; idx < 16 * 64; idx += 32) {
        int t = idx >> 6, d = idx & 63;
        long long rowg = (long long)(b * 16 + t) * 768;
        qs[t][d]  = f2bf(qkv[rowg + h * 64 + d]);
        ks[t][d]  = f2bf(qkv[rowg + 256 + h * 64 + d]);
        vT[d][t]  = f2bf(qkv[rowg + 512 + h * 64 + d]);
    }
    for (int idx = lane; idx < 64 * 16; idx += 32) {
        int d = idx >> 4, s = idx & 15;
        vT[d][16 + s] = 0;
    }
    __syncthreads();

    // scores = q(16x64) @ k^T(64x16), two K=32 WMMAs. B layout: Bs[n=s][k=d] == ks natural.
    v8f accS = {0.f,0.f,0.f,0.f,0.f,0.f,0.f,0.f};
#pragma unroll
    for (int kb = 0; kb < 2; ++kb) {
        FragU a, bb;
        int kofA = kb * 32 + (halfsel << 4);
        a.h[0] = *(const ush8*)&qs[l15][kofA];
        a.h[1] = *(const ush8*)&qs[l15][kofA + 8];
        int kofB = kb * 32 + (halfsel << 4);
        bb.h[0] = *(const ush8*)&ks[l15][kofB];
        bb.h[1] = *(const ush8*)&ks[l15][kofB + 8];
        accS = __builtin_amdgcn_wmma_f32_16x16x32_bf16(
            false, a.bf, false, bb.bf, (short)0, accS, false, false);
    }
#pragma unroll
    for (int r = 0; r < 8; ++r)
        sc[r + (halfsel << 3)][l15] = accS[r] * 0.125f;   // 1/sqrt(64)
    __syncthreads();

    // rowwise softmax: lanes 0..15 each own one row
    if (lane < 16) {
        float mx = -1e30f;
        for (int c = 0; c < 16; ++c) mx = fmaxf(mx, sc[lane][c]);
        float e[16], sum = 0.f;
        for (int c = 0; c < 16; ++c) { e[c] = __expf(sc[lane][c] - mx); sum += e[c]; }
        float inv = __builtin_amdgcn_rcpf(sum);
        for (int c = 0; c < 16; ++c) Ps[lane][c] = f2bf(e[c] * inv);
        for (int c = 16; c < 32; ++c) Ps[lane][c] = 0;
    }
    __syncthreads();

    // out = P(16x32pad) @ v(32pad x 64): 4 N-tiles
    FragU ap;
    {
        int kof = halfsel << 4;
        ap.h[0] = *(const ush8*)&Ps[l15][kof];
        ap.h[1] = *(const ush8*)&Ps[l15][kof + 8];
    }
#pragma unroll
    for (int nt = 0; nt < 4; ++nt) {
        FragU bv;
        int n = nt * 16 + l15;
        bv.h[0] = *(const ush8*)&vT[n][halfsel << 4];
        bv.h[1] = *(const ush8*)&vT[n][(halfsel << 4) + 8];
        v8f accO = {0.f,0.f,0.f,0.f,0.f,0.f,0.f,0.f};
        accO = __builtin_amdgcn_wmma_f32_16x16x32_bf16(
            false, ap.bf, false, bv.bf, (short)0, accO, false, false);
#pragma unroll
        for (int r = 0; r < 8; ++r) {
            int t = r + (halfsel << 3);
            int d = nt * 16 + l15;
            attended[(long long)(b * 16 + t) * 256 + h * 64 + d] = f2bf(accO[r]);
        }
    }
}

// ---------------- LayerNorm over D=256 after mean over T (T=1 for plain LN) ----------------
__global__ __launch_bounds__(256) void meanT_ln(
    const float* __restrict__ x, const float* __restrict__ g, const float* __restrict__ be,
    float* __restrict__ outF, unsigned short* __restrict__ outB, int T)
{
    const int b = blockIdx.x;
    const int c = threadIdx.x;
    float s = 0.f;
    for (int t = 0; t < T; ++t) s += x[((long long)b * T + t) * 256 + c];
    s *= (1.0f / T);
    __shared__ float red[256];
    red[c] = s; __syncthreads();
    for (int off = 128; off > 0; off >>= 1) { if (c < off) red[c] += red[c + off]; __syncthreads(); }
    float mu = red[0] * (1.0f / 256.0f); __syncthreads();
    float d = s - mu;
    red[c] = d * d; __syncthreads();
    for (int off = 128; off > 0; off >>= 1) { if (c < off) red[c] += red[c + off]; __syncthreads(); }
    float var = red[0] * (1.0f / 256.0f);
    float y = d * rsqrtf(var + 1e-5f) * g[c] + be[c];
    long long o = (long long)b * 256 + c;
    if (outF) outF[o] = y;
    if (outB) outB[o] = f2bf(y);
}

// ---------------- ODE helpers ----------------
__global__ __launch_bounds__(256) void init_z(const float* __restrict__ obs,
                                              float* __restrict__ z, float* __restrict__ pred0)
{
    int idx = blockIdx.x * blockDim.x + threadIdx.x;   // 256*256
    if (idx >= 256 * 256) return;
    int b = idx >> 8, c = idx & 255;
    float v = obs[(long long)(b * 16) * 256 + c];      // obs[:,0,:]
    z[idx] = v;
    pred0[(long long)b * 4096 + c] = v;                // pred[:,0,:]
}

__global__ __launch_bounds__(32) void te_kernel(
    const float* __restrict__ tsv, int i, int j,
    const float* __restrict__ Wt1, const float* __restrict__ bt1,
    const float* __restrict__ Wt2, const float* __restrict__ bt2,
    float* __restrict__ te)
{
    int lane = threadIdx.x;
    float t0 = tsv[i];
    float dt = (tsv[i + 1] - t0) * 0.1f;
    float t = t0 + (float)j * dt;
    __shared__ float gsh[32];
    gsh[lane] = gelu_f(t * Wt1[lane] + bt1[lane]);
    __syncthreads();
    float s = 0.f;
    for (int k = 0; k < 32; ++k) s += gsh[k] * Wt2[k * 32 + lane];
    te[lane] = s + bt2[lane];
}

__global__ __launch_bounds__(256) void build_xt(const float* __restrict__ z,
                                                const float* __restrict__ te,
                                                unsigned short* __restrict__ xt)
{
    int idx = blockIdx.x * blockDim.x + threadIdx.x;   // 256*288
    if (idx >= 256 * 288) return;
    int r = idx / 288, c = idx % 288;
    xt[idx] = f2bf(c < 256 ? z[r * 256 + c] : te[c - 256]);
}

__global__ __launch_bounds__(32) void ode_update(
    float* __restrict__ z, const float* __restrict__ D3,
    const float* __restrict__ tsv, int i, float* __restrict__ predSlot)
{
    int b = blockIdx.x, lane = threadIdx.x;
    float dd = (tsv[i + 1] - tsv[i]) * 0.1f;
    float s = 0.f;
    for (int c = lane; c < 256; c += 32) s += D3[b * 256 + c];
#pragma unroll
    for (int m = 16; m > 0; m >>= 1) s += __shfl_xor(s, m, 32);
    float mean = s * (1.0f / 256.0f);
    for (int c = lane; c < 256; c += 32) {
        float nz = z[b * 256 + c] + dd * (D3[b * 256 + c] - mean);
        z[b * 256 + c] = nz;
        if (predSlot) predSlot[(long long)b * 4096 + c] = nz;
    }
}

// ---------------- scoring path ----------------
__global__ __launch_bounds__(32) void meandev_kernel(const float* __restrict__ pred,
                                                     const float* __restrict__ obs,
                                                     float* __restrict__ md)
{
    int b = blockIdx.x, lane = threadIdx.x;
    float s = 0.f;
    for (int idx = lane; idx < 16 * 256; idx += 32) {
        float d = pred[(long long)b * 4096 + idx] - obs[(long long)b * 4096 + idx];
        s += d * d;
    }
#pragma unroll
    for (int m = 16; m > 0; m >>= 1) s += __shfl_xor(s, m, 32);
    if (lane == 0) md[b] = s * (1.0f / 4096.0f);
}

__global__ __launch_bounds__(256) void stats_kernel(const float* __restrict__ pred,
                                                    const float* __restrict__ obs,
                                                    unsigned short* __restrict__ stats)
{
    int idx = blockIdx.x * blockDim.x + threadIdx.x;   // 256*512
    if (idx >= 256 * 512) return;
    int b = idx >> 9, c = idx & 511;
    const float* src = (c < 256) ? pred : obs;
    int cc = c & 255;
    float s = 0.f;
    for (int t = 0; t < 16; ++t) s += src[(long long)b * 4096 + t * 256 + cc];
    stats[idx] = f2bf(s * (1.0f / 16.0f));
}

__global__ __launch_bounds__(32) void score_kernel(const float* __restrict__ S1,
                                                   const float* __restrict__ Wa2,
                                                   const float* __restrict__ ba2,
                                                   const float* __restrict__ md,
                                                   float* __restrict__ score)
{
    int b = blockIdx.x, lane = threadIdx.x;
    float s = 0.f;
    for (int c = lane; c < 512; c += 32) s += S1[b * 512 + c] * Wa2[c];
#pragma unroll
    for (int m = 16; m > 0; m >>= 1) s += __shfl_xor(s, m, 32);
    if (lane == 0) score[b] = s + ba2[0] + md[b];
}

// ---------------- orchestration ----------------
extern "C" void kernel_launch(void* const* d_in, const int* in_sizes, int n_in,
                              void* d_out, int out_size, void* d_ws, size_t ws_size,
                              hipStream_t stream)
{
    (void)in_sizes; (void)n_in; (void)out_size; (void)ws_size;
    const float* clr     = (const float*)d_in[0];
    const float* tsv     = (const float*)d_in[1];
    const float* W_enc1  = (const float*)d_in[2];
    const float* b_enc1  = (const float*)d_in[3];
    const float* W_enc2  = (const float*)d_in[4];
    const float* b_enc2  = (const float*)d_in[5];
    const float* ln_enc_g= (const float*)d_in[6];
    const float* ln_enc_b= (const float*)d_in[7];
    const float* Wt1     = (const float*)d_in[8];
    const float* bt1     = (const float*)d_in[9];
    const float* Wt2     = (const float*)d_in[10];
    const float* bt2     = (const float*)d_in[11];
    const float* Wv1     = (const float*)d_in[12];
    const float* bv1     = (const float*)d_in[13];
    const float* Wv2     = (const float*)d_in[14];
    const float* bv2     = (const float*)d_in[15];
    const float* Wv3     = (const float*)d_in[16];
    const float* bv3     = (const float*)d_in[17];
    const float* W_qkv   = (const float*)d_in[18];
    const float* b_qkv   = (const float*)d_in[19];
    const float* W_o     = (const float*)d_in[20];
    const float* b_o     = (const float*)d_in[21];
    const float* ln_tr_g = (const float*)d_in[22];
    const float* ln_tr_b = (const float*)d_in[23];
    const float* Wa1     = (const float*)d_in[24];
    const float* ba1     = (const float*)d_in[25];
    const float* Wa2     = (const float*)d_in[26];
    const float* ba2     = (const float*)d_in[27];

    float* out = (float*)d_out;
    float* out_emb  = out;                 // 256*256
    float* out_scr  = out + 65536;         // 256
    float* out_pred = out + 65792;         // 256*16*256
    float* out_obs  = out + 1114368;       // 256*16*256

    // workspace bump allocator
    size_t off = 0;
    char* ws = (char*)d_ws;
    auto alloc = [&](size_t bytes) -> void* {
        void* p = ws + off;
        off += (bytes + 255) & ~(size_t)255;
        return p;
    };
    const int KP1 = 5024;  // 5000 padded to 32
    unsigned short* Xbf     = (unsigned short*)alloc((size_t)4096 * KP1 * 2);
    unsigned short* W1T     = (unsigned short*)alloc((size_t)512 * KP1 * 2);
    unsigned short* Hbf     = (unsigned short*)alloc((size_t)4096 * 512 * 2);
    unsigned short* W2T     = (unsigned short*)alloc((size_t)256 * 512 * 2);
    float*          obs_pre = (float*)alloc((size_t)4096 * 256 * 4);
    unsigned short* obs_bf  = (unsigned short*)alloc((size_t)4096 * 256 * 2);
    unsigned short* WqkvT   = (unsigned short*)alloc((size_t)768 * 256 * 2);
    float*          qkv     = (float*)alloc((size_t)4096 * 768 * 4);
    unsigned short* attbf   = (unsigned short*)alloc((size_t)4096 * 256 * 2);
    unsigned short* WoT     = (unsigned short*)alloc((size_t)256 * 256 * 2);
    float*          att_out = (float*)alloc((size_t)4096 * 256 * 4);
    float*          z       = (float*)alloc((size_t)256 * 256 * 4);
    float*          te      = (float*)alloc(32 * 4);
    unsigned short* XTbf    = (unsigned short*)alloc((size_t)256 * 288 * 2);
    unsigned short* Wv1T    = (unsigned short*)alloc((size_t)512 * 288 * 2);
    unsigned short* D1bf    = (unsigned short*)alloc((size_t)256 * 512 * 2);
    unsigned short* Wv2T    = (unsigned short*)alloc((size_t)512 * 512 * 2);
    unsigned short* D2bf    = (unsigned short*)alloc((size_t)256 * 512 * 2);
    unsigned short* Wv3T    = (unsigned short*)alloc((size_t)256 * 512 * 2);
    float*          D3      = (float*)alloc((size_t)256 * 256 * 4);
    float*          md      = (float*)alloc(256 * 4);
    unsigned short* statsbf = (unsigned short*)alloc((size_t)256 * 512 * 2);
    unsigned short* Wa1T    = (unsigned short*)alloc((size_t)512 * 512 * 2);
    float*          S1      = (float*)alloc((size_t)256 * 512 * 4);

    auto blks = [](long long n) { return (unsigned)((n + 255) / 256); };

    // ---- weight/activation conversions (weights pre-transposed to N x Kp) ----
    cvt_act_pad<<<blks((long long)4096 * KP1), 256, 0, stream>>>(clr, Xbf, 4096, 5000, KP1);
    cvt_wT_pad<<<blks((long long)512 * KP1), 256, 0, stream>>>(W_enc1, W1T, 5000, 512, KP1);
    cvt_wT_pad<<<blks(256 * 512), 256, 0, stream>>>(W_enc2, W2T, 512, 256, 512);
    cvt_wT_pad<<<blks(768 * 256), 256, 0, stream>>>(W_qkv, WqkvT, 256, 768, 256);
    cvt_wT_pad<<<blks(256 * 256), 256, 0, stream>>>(W_o, WoT, 256, 256, 256);
    cvt_wT_pad<<<blks(512 * 288), 256, 0, stream>>>(Wv1, Wv1T, 288, 512, 288);
    cvt_wT_pad<<<blks(512 * 512), 256, 0, stream>>>(Wv2, Wv2T, 512, 512, 512);
    cvt_wT_pad<<<blks(256 * 512), 256, 0, stream>>>(Wv3, Wv3T, 512, 256, 512);
    cvt_wT_pad<<<blks(512 * 512), 256, 0, stream>>>(Wa1, Wa1T, 512, 512, 512);

    // ---- encoder ----
    gemm_bf16_wmma<1, false, true><<<dim3(512 / BN, 4096 / BM), 256, 0, stream>>>(
        Xbf, W1T, b_enc1, nullptr, Hbf, 4096, 512, KP1);
    gemm_bf16_wmma<0, true, false><<<dim3(256 / BN, 4096 / BM), 256, 0, stream>>>(
        Hbf, W2T, b_enc2, obs_pre, nullptr, 4096, 256, 512);
    meanT_ln<<<4096, 256, 0, stream>>>(obs_pre, ln_enc_g, ln_enc_b, out_obs, obs_bf, 1);

    // ---- attention path ----
    gemm_bf16_wmma<0, true, false><<<dim3(768 / BN, 4096 / BM), 256, 0, stream>>>(
        obs_bf, WqkvT, b_qkv, qkv, nullptr, 4096, 768, 256);
    attn_kernel<<<1024, 32, 0, stream>>>(qkv, attbf);
    gemm_bf16_wmma<0, true, false><<<dim3(256 / BN, 4096 / BM), 256, 0, stream>>>(
        attbf, WoT, b_o, att_out, nullptr, 4096, 256, 256);
    meanT_ln<<<256, 256, 0, stream>>>(att_out, ln_tr_g, ln_tr_b, out_emb, nullptr, 16);

    // ---- Neural ODE integration: 15 intervals x 10 steps ----
    init_z<<<blks(256 * 256), 256, 0, stream>>>(out_obs, z, out_pred);
    for (int i = 0; i < 15; ++i) {
        for (int j = 0; j < 10; ++j) {
            te_kernel<<<1, 32, 0, stream>>>(tsv, i, j, Wt1, bt1, Wt2, bt2, te);
            build_xt<<<blks(256 * 288), 256, 0, stream>>>(z, te, XTbf);
            gemm_bf16_wmma<1, false, true><<<dim3(512 / BN, 256 / BM), 256, 0, stream>>>(
                XTbf, Wv1T, bv1, nullptr, D1bf, 256, 512, 288);
            gemm_bf16_wmma<1, false, true><<<dim3(512 / BN, 256 / BM), 256, 0, stream>>>(
                D1bf, Wv2T, bv2, nullptr, D2bf, 256, 512, 512);
            gemm_bf16_wmma<0, true, false><<<dim3(256 / BN, 256 / BM), 256, 0, stream>>>(
                D2bf, Wv3T, bv3, D3, nullptr, 256, 256, 512);
            float* predSlot = (j == 9) ? (out_pred + (size_t)(i + 1) * 256) : nullptr;
            ode_update<<<256, 32, 0, stream>>>(z, D3, tsv, i, predSlot);
        }
    }

    // ---- scoring ----
    meandev_kernel<<<256, 32, 0, stream>>>(out_pred, out_obs, md);
    stats_kernel<<<blks(256 * 512), 256, 0, stream>>>(out_pred, out_obs, statsbf);
    gemm_bf16_wmma<1, true, false><<<dim3(512 / BN, 256 / BM), 256, 0, stream>>>(
        statsbf, Wa1T, ba1, S1, nullptr, 256, 512, 512);
    score_kernel<<<256, 32, 0, stream>>>(S1, Wa2, ba2, md, out_scr);
}
